// MultiHeadAttention_68779606278372
// MI455X (gfx1250) — compile-verified, matrix operand order checked by reading, not run on hardware
//
#include <hip/hip_runtime.h>

// MHA for MI455X (gfx1250, wave32, WMMA + TDM).
// S=2048 B=4 H=1024 NH=16 DK=64. All GEMMs via v_wmma_f32_16x16x32_bf16.

#define S_LEN 2048
#define BATCH 4
#define HDIM  1024
#define NHEAD 16
#define DKD   64
#define MROWS (S_LEN * BATCH)   // 8192 flattened (s,b) rows

typedef __attribute__((ext_vector_type(16))) __bf16 v16bf;
typedef __attribute__((ext_vector_type(8)))  __bf16 bf16x8;
typedef __attribute__((ext_vector_type(8)))  float  v8f;
typedef __attribute__((ext_vector_type(4)))  float  f32x4;

#if defined(__has_builtin)
#if __has_builtin(__builtin_amdgcn_tensor_load_to_lds) && \
    __has_builtin(__builtin_amdgcn_s_wait_tensorcnt)
#define USE_TDM 1
#endif
#endif

// ds_swizzle_b32 XOR-butterfly step (group-of-32 encoding: offset[15]=0,
// xor_mask=offset[14:10], and_mask=0x1F). XOR of 1/2/4/8 stays inside each
// 16-lane half, matching the C-fragment row layout.
#define SWZ_XOR(v, pat)                                                        \
  __builtin_bit_cast(float, __builtin_amdgcn_ds_swizzle(                       \
                                __builtin_bit_cast(int, (v)), (pat)))
#define RED16(v, OPX)                                                          \
  do {                                                                         \
    v = OPX(v, SWZ_XOR(v, 0x201F)); /* xor 8 */                                \
    v = OPX(v, SWZ_XOR(v, 0x101F)); /* xor 4 */                                \
    v = OPX(v, SWZ_XOR(v, 0x081F)); /* xor 2 */                                \
    v = OPX(v, SWZ_XOR(v, 0x041F)); /* xor 1 */                                \
  } while (0)
#define FADD(a, b) ((a) + (b))

__device__ __forceinline__ __bf16 f2bf(float f) {
  unsigned u = __builtin_bit_cast(unsigned, f);
  u += 0x7FFFu + ((u >> 16) & 1u);              // round-to-nearest-even
  unsigned short h = (unsigned short)(u >> 16);
  return __builtin_bit_cast(__bf16, h);
}

// 16-bit A/B fragment, 16x32 (MxK) / 32x16 (KxN) per ISA layout:
// lane l: row/col = l&15 ; half kh=(l>>4) selects K chunks {kh*8..kh*8+7} and
// {16+kh*8..16+kh*8+7}. x (MxK) and W (NxK) are both K-contiguous, so one
// loader serves A and B fragments.
__device__ __forceinline__ v16bf frag_bf16(const __bf16* p, int kh) {
  bf16x8 lo = *(const bf16x8*)(p + kh * 8);
  bf16x8 hi = *(const bf16x8*)(p + 16 + kh * 8);
  return __builtin_shufflevector(lo, hi, 0,1,2,3,4,5,6,7,8,9,10,11,12,13,14,15);
}

__device__ __forceinline__ v16bf frag_f32(const float* p, int kh) {
  const f32x4* q0 = (const f32x4*)(p + kh * 8);
  const f32x4* q1 = (const f32x4*)(p + 16 + kh * 8);
  f32x4 a0 = q0[0], a1 = q0[1], b0 = q1[0], b1 = q1[1];
  v16bf r;
#pragma unroll
  for (int i = 0; i < 4; ++i) {
    r[i]      = f2bf(a0[i]);
    r[4 + i]  = f2bf(a1[i]);
    r[8 + i]  = f2bf(b0[i]);
    r[12 + i] = f2bf(b1[i]);
  }
  return r;
}

__device__ __forceinline__ v8f wmma_bf16(v16bf a, v16bf b, v8f c) {
  return __builtin_amdgcn_wmma_f32_16x16x32_bf16(false, a, false, b, (short)0, c,
                                                 false, false);
}

#ifdef USE_TDM
typedef __attribute__((ext_vector_type(4))) unsigned tdm_u32x4;
typedef __attribute__((ext_vector_type(8))) int      tdm_i32x8;
typedef __attribute__((ext_vector_type(4))) int      tdm_i32x4;

// 2-D TDM tile load, bf16 elements. Dims/strides in elements.
// D# packing per CDNA5 ISA.
// This toolchain exposes the 6-arg builtin:
//   (u32x4 g0, i32x8 g1, i32x4 g2, i32x4 g3, i32x8 extra, i32 cpol)
__device__ __forceinline__ void tdm_load_2d(unsigned lds_off, const void* gptr,
                                            unsigned tile0, unsigned tile1,
                                            unsigned tdim0, unsigned tdim1,
                                            unsigned long long stride0) {
  unsigned long long ga = (unsigned long long)gptr;
  tdm_u32x4 g0;
  g0[0] = 1u;                                      // count=1, user mode
  g0[1] = lds_off;                                 // lds_addr[31:0]
  g0[2] = (unsigned)ga;                            // global_addr[31:0]
  g0[3] = (unsigned)((ga >> 32) & 0x01FFFFFFu)     // global_addr[56:32]
        | (2u << 30);                              // type=2 ("image")
  tdm_i32x8 g1;
  g1[0] = (int)(1u << 16);                         // data_size=1 (2B), wg_mask=0
  g1[1] = (int)((tdim0 & 0xFFFFu) << 16);          // tensor_dim0[15:0]
  g1[2] = (int)(((tdim0 >> 16) & 0xFFFFu) | ((tdim1 & 0xFFFFu) << 16));
  g1[3] = (int)(((tdim1 >> 16) & 0xFFFFu) | ((tile0 & 0xFFFFu) << 16));
  g1[4] = (int)(tile1 & 0xFFFFu);                  // tile_dim1; tile_dim2=0
  g1[5] = (int)(unsigned)(stride0 & 0xFFFFFFFFu);  // dim0_stride[31:0]
  g1[6] = (int)(unsigned)((stride0 >> 32) & 0xFFFFu);
  g1[7] = 0;
  tdm_i32x4 gz4 = {0, 0, 0, 0};                    // 2-D: groups 2/3 unused
  tdm_i32x8 gz8 = {0, 0, 0, 0, 0, 0, 0, 0};
  __builtin_amdgcn_tensor_load_to_lds(g0, g1, gz4, gz4, gz8, 0);
}
#endif

// ---------------------------------------------------------------------------
// GEMM: Out = A(M x K) * W(N x K)^T   (nn.Linear, bias=false)
// MODE 0: Out bf16, head-major [b,h,s,d]        (Q, K projections)
// MODE 1: Out bf16, transposed [b,h,d,s]        (V projection)
// MODE 2: Out f32,  row-major  [(s,b), H]       (final O projection -> d_out)
// ---------------------------------------------------------------------------
template <int MODE, bool AF32>
__global__ __launch_bounds__(256) void gemm_wmma(const void* __restrict__ Av,
                                                 const float* __restrict__ W,
                                                 void* __restrict__ Outv) {
  const int lane = threadIdx.x & 31;
  const int wave = threadIdx.x >> 5;
  const int n16  = lane & 15;
  const int kh   = (lane >> 4) & 1;
  const int rowBase = blockIdx.y * 128 + wave * 16;
  const int colBase = blockIdx.x * 64;
  const int K = HDIM;

  const float*  Af = (const float*)Av;
  const __bf16* Ab = (const __bf16*)Av;

  v8f acc[4] = {};

  for (int k0 = 0; k0 < K; k0 += 32) {
    v16bf a;
    if constexpr (AF32) {
      const float* ap = Af + (size_t)(rowBase + n16) * K + k0;
      a = frag_f32(ap, kh);
      if (k0 + 32 < K) __builtin_prefetch(ap + 32, 0, 1);   // global_prefetch_b8
    } else {
      const __bf16* ap = Ab + (size_t)(rowBase + n16) * K + k0;
      a = frag_bf16(ap, kh);
      if (k0 + 32 < K) __builtin_prefetch(ap + 32, 0, 1);
    }
#pragma unroll
    for (int t = 0; t < 4; ++t) {
      v16bf bfr = frag_f32(W + (size_t)(colBase + t * 16 + n16) * K + k0, kh);
      acc[t] = wmma_bf16(a, bfr, acc[t]);
    }
  }

  // C layout: VGPR r, lane l -> row rowBase + r + 8*(l>>4), col colBase+t*16+(l&15)
#pragma unroll
  for (int r = 0; r < 8; ++r) {
    const int rowg = rowBase + r + 8 * kh;        // (s,b) flat: row = s*B + b
    const int s = rowg / BATCH, b = rowg % BATCH;
#pragma unroll
    for (int t = 0; t < 4; ++t) {
      const int col = colBase + t * 16 + n16;     // j = h*DK + d
      const float v = acc[t][r];
      if constexpr (MODE == 0) {
        const int h = col >> 6, d = col & 63;
        ((__bf16*)Outv)[(((size_t)(b * NHEAD + h)) * S_LEN + s) * DKD + d] = f2bf(v);
      } else if constexpr (MODE == 1) {
        const int h = col >> 6, d = col & 63;
        ((__bf16*)Outv)[(((size_t)(b * NHEAD + h)) * DKD + d) * S_LEN + s] = f2bf(v);
      } else {
        ((float*)Outv)[(size_t)rowg * HDIM + col] = v;
      }
    }
  }
}

// ---------------------------------------------------------------------------
// Flash attention: grid (S/64, B*NH), 128 threads = 4 waves, 16 queries/wave.
// K/V tiles for the block's (b,h) are staged in LDS by the Tensor Data Mover
// (double-buffered, overlapped with compute); all 4 waves share them.
// Per 32-key step: 4 WMMA (scores) + online softmax + 4 WMMA (P*V).
// ---------------------------------------------------------------------------
__global__ __launch_bounds__(128) void attn_wmma(const __bf16* __restrict__ Qh,
                                                 const __bf16* __restrict__ Kh,
                                                 const __bf16* __restrict__ Vt,
                                                 const unsigned char* __restrict__ pad,
                                                 __bf16* __restrict__ AO) {
#ifdef USE_TDM
  __shared__ __align__(16) __bf16 Kb[2][32][DKD];    // 2 x 4KB  [key][d]
  __shared__ __align__(16) __bf16 Vb[2][DKD][32];    // 2 x 4KB  [d][key]
#endif
  __shared__ __align__(16) __bf16 plds[4][16][32];   // P bounce, 1KB/wave

  const int lane = threadIdx.x & 31;
  const int wave = threadIdx.x >> 5;
  const int n16  = lane & 15;
  const int kh   = (lane >> 4) & 1;
  const int bh = blockIdx.y;
  const int b  = bh / NHEAD;
  const int h  = bh % NHEAD;
  const int q0 = (blockIdx.x * 4 + wave) * 16;
  const float kscale = 0.125f;                       // 1/sqrt(DK)

  // Q tile A-fragments (d 0..31 and 32..63), resident for the whole loop.
  const __bf16* qrow = Qh + ((size_t)bh * S_LEN + q0 + n16) * DKD;
  const v16bf aq0 = frag_bf16(qrow, kh);
  const v16bf aq1 = frag_bf16(qrow + 32, kh);

  v8f o[4] = {};
  float mrow[8], lrow[8];
#pragma unroll
  for (int r = 0; r < 8; ++r) { mrow[r] = -3.0e38f; lrow[r] = 0.0f; }

#ifdef USE_TDM
  const __bf16* Kbase = Kh + (size_t)bh * S_LEN * DKD;   // [s][d]
  const __bf16* Vbase = Vt + (size_t)bh * DKD * S_LEN;   // [d][s]
  if (wave == 0) {       // prologue: DMA tiles for kb=0 into buffer 0
    tdm_load_2d((unsigned)(unsigned long long)&Kb[0][0][0], Kbase,
                /*tile*/ DKD, 32, /*tensor*/ DKD, S_LEN, DKD);
    tdm_load_2d((unsigned)(unsigned long long)&Vb[0][0][0], Vbase,
                /*tile*/ 32, DKD, /*tensor*/ S_LEN, DKD, S_LEN);
  }
  int cur = 0;
#endif

  for (int kb = 0; kb < S_LEN; kb += 32) {
#ifdef USE_TDM
    if (wave == 0) __builtin_amdgcn_s_wait_tensorcnt(0);  // tiles landed in LDS
    __syncthreads();                                      // publish to block
    if (wave == 0 && kb + 32 < S_LEN) {                   // prefetch next tiles
      tdm_load_2d((unsigned)(unsigned long long)&Kb[cur ^ 1][0][0],
                  Kbase + (size_t)(kb + 32) * DKD, DKD, 32, DKD, S_LEN, DKD);
      tdm_load_2d((unsigned)(unsigned long long)&Vb[cur ^ 1][0][0],
                  Vbase + (kb + 32), 32, DKD, S_LEN, DKD, S_LEN);
    }
#endif
    // ---- scores tile 16x32 (two 16x16 column tiles) ----
    float z[2][8];
#pragma unroll
    for (int nt = 0; nt < 2; ++nt) {
      const int krow = kb + nt * 16 + n16;
      const bool km = pad[(size_t)krow * BATCH + b] != 0;   // PAD key
#ifdef USE_TDM
      const __bf16* kp = &Kb[cur][nt * 16 + n16][0];
#else
      const __bf16* kp = Kh + ((size_t)bh * S_LEN + krow) * DKD;
#endif
      v8f sacc = {};
      sacc = wmma_bf16(aq0, frag_bf16(kp, kh), sacc);
      sacc = wmma_bf16(aq1, frag_bf16(kp + 32, kh), sacc);
#pragma unroll
      for (int r = 0; r < 8; ++r)
        z[nt][r] = km ? -3.0e38f : sacc[r] * kscale;        // mask then scale
    }

    // ---- online softmax; row = r + 8*kh, cols live across lanes (l&15) ----
#pragma unroll
    for (int r = 0; r < 8; ++r) {
      float tm = fmaxf(z[0][r], z[1][r]);
      RED16(tm, fmaxf);
      const float mn   = fmaxf(mrow[r], tm);
      const float corr = __expf(mrow[r] - mn);
      const float p0 = __expf(z[0][r] - mn);
      const float p1 = __expf(z[1][r] - mn);
      float ts = p0 + p1;
      RED16(ts, FADD);
      lrow[r] = lrow[r] * corr + ts;
      mrow[r] = mn;
#pragma unroll
      for (int t = 0; t < 4; ++t) o[t][r] *= corr;
      plds[wave][r + 8 * kh][n16]      = f2bf(p0);          // C-layout -> LDS
      plds[wave][r + 8 * kh][16 + n16] = f2bf(p1);
    }
    asm volatile("s_wait_dscnt 0" ::: "memory");            // store->load, same wave

    // ---- P (16x32) * V (32x64): A from LDS, B = V^T tile ----
    const v16bf ap = frag_bf16(&plds[wave][n16][0], kh);
#pragma unroll
    for (int t = 0; t < 4; ++t) {
#ifdef USE_TDM
      const __bf16* vp = &Vb[cur][t * 16 + n16][0];
#else
      const __bf16* vp = Vt + ((size_t)bh * DKD + t * 16 + n16) * S_LEN + kb;
#endif
      o[t] = wmma_bf16(ap, frag_bf16(vp, kh), o[t]);
    }
#ifdef USE_TDM
    cur ^= 1;
#endif
  }

  // ---- normalize, zero PAD query rows, store bf16 row-major [(s,b), H] ----
#pragma unroll
  for (int r = 0; r < 8; ++r) {
    const int s = q0 + r + 8 * kh;
    const bool qm = pad[(size_t)s * BATCH + b] != 0;
    const float inv = 1.0f / lrow[r];
    const size_t base = ((size_t)s * BATCH + b) * HDIM + h * DKD;
#pragma unroll
    for (int t = 0; t < 4; ++t) {
      const float v = qm ? 0.0f : o[t][r] * inv;
      AO[base + t * 16 + n16] = f2bf(v);
    }
  }
}

// ---------------------------------------------------------------------------
extern "C" void kernel_launch(void* const* d_in, const int* in_sizes, int n_in,
                              void* d_out, int out_size, void* d_ws, size_t ws_size,
                              hipStream_t stream) {
  (void)in_sizes; (void)n_in; (void)out_size; (void)ws_size;
  const float* value = (const float*)d_in[0];
  const float* key_t = (const float*)d_in[1];
  const float* query = (const float*)d_in[2];
  const unsigned char* pad = (const unsigned char*)d_in[3];  // jnp bool (S,B)
  const float* Wq = (const float*)d_in[4];
  const float* Wk = (const float*)d_in[5];
  const float* Wv = (const float*)d_in[6];
  const float* Wo = (const float*)d_in[7];
  float* out = (float*)d_out;

  const size_t nElem = (size_t)MROWS * HDIM;   // 8.39M elements per tensor
  __bf16* Qh = (__bf16*)d_ws;                  // [b,h,s,d]
  __bf16* Kh = Qh + nElem;                     // [b,h,s,d]
  __bf16* Vt = Kh + nElem;                     // [b,h,d,s]
  __bf16* AO = Vt + nElem;                     // [(s,b), H]  (total 64 MB ws)

  const dim3 blk(256);
  const dim3 grd(HDIM / 64, MROWS / 128);      // (16, 64)

  gemm_wmma<0, true ><<<grd, blk, 0, stream>>>(query, Wq, Qh);
  gemm_wmma<0, true ><<<grd, blk, 0, stream>>>(key_t, Wk, Kh);
  gemm_wmma<1, true ><<<grd, blk, 0, stream>>>(value, Wv, Vt);
  attn_wmma<<<dim3(S_LEN / 64, BATCH * NHEAD), dim3(128), 0, stream>>>(Qh, Kh, Vt,
                                                                       pad, AO);
  gemm_wmma<2, false><<<grd, blk, 0, stream>>>(AO, Wo, out);
}
